// DeslicingDecoder_23570780520661
// MI455X (gfx1250) — compile-verified, matrix operand order checked by reading, not run on hardware
//
#include <hip/hip_runtime.h>
#include <hip/hip_bf16.h>
#include <math.h>

typedef __bf16 bf16_t;
typedef __attribute__((ext_vector_type(16))) __bf16 v16bf;
typedef __attribute__((ext_vector_type(8)))  float  v8f;

#define M_TILE 32

// ---------------------------------------------------------------------------
// K0: pack a 256x256 f32 weight (row-major [out_j][in_d]) into bf16 WMMA
// B-fragment layout: fragment (nt,kt) = 16 cols x 32 ks; per lane 16 bf16
// contiguous (32B) so GEMM loads are two global_load_b128 per fragment.
// ISA 16-bit layout: lane = (n%16) + 16*b ; element i -> k = 8*b + (i<8?i:i+8)
// ---------------------------------------------------------------------------
__global__ void pack_weight_bf16(const float* __restrict__ W, bf16_t* __restrict__ P) {
    int t    = blockIdx.x * blockDim.x + threadIdx.x;   // 65536 threads total
    int i    = t & 15;
    int lane = (t >> 4) & 31;
    int frag = t >> 9;            // nt*8 + kt, 0..127
    int kt   = frag & 7;
    int nt   = frag >> 3;
    int j    = nt * 16 + (lane & 15);
    int b    = lane >> 4;
    int k    = kt * 32 + b * 8 + (i < 8 ? i : i + 8);
    P[t] = (bf16_t)W[j * 256 + k];
}

// ---------------------------------------------------------------------------
// K1: tok3 = evolved_tokens @ Wp.T + bp   (256x256x256, tiny -> VALU)
// ---------------------------------------------------------------------------
__global__ void proj_kernel(const float* __restrict__ E, const float* __restrict__ Wp,
                            const float* __restrict__ bp, float* __restrict__ tok) {
    __shared__ float row[256];
    int r = blockIdx.x, j = threadIdx.x;
    row[j] = E[r * 256 + j];
    __syncthreads();
    float acc = bp[j];
    #pragma unroll 4
    for (int d = 0; d < 256; ++d) acc = fmaf(row[d], Wp[j * 256 + d], acc);
    tok[r * 256 + j] = acc;
}

// ---------------------------------------------------------------------------
// K2: z_out = LN(einsum(attn, tok3[var_batch]) + z_var_0); masks/offsets/ranges
// One block (256 threads) per variable row. tok3 is 256KB -> L2 resident.
// ---------------------------------------------------------------------------
__global__ void gather_ln_kernel(const int* __restrict__ vb, const float* __restrict__ attn,
                                 const int* __restrict__ vt, const float* __restrict__ zvar,
                                 const float* __restrict__ vfeat, const float* __restrict__ tok,
                                 const float* __restrict__ fng, const float* __restrict__ fnb,
                                 float* __restrict__ z_out,
                                 float* __restrict__ o_mb, float* __restrict__ o_mis,
                                 float* __restrict__ o_mil, float* __restrict__ o_off,
                                 float* __restrict__ o_rng, int N) {
    int n = blockIdx.x, d = threadIdx.x;
    __shared__ float attn_s[32];
    __shared__ float partS[8], partQ[8];
    __shared__ float s_mean, s_rstd;
    int b = vb[n];
    if (d < 32) attn_s[d] = attn[(size_t)n * 32 + d];
    __syncthreads();
    const float* base = tok + (size_t)b * 32 * 256 + d;
    float acc = zvar[(size_t)n * 256 + d];
    #pragma unroll
    for (int k = 0; k < 32; ++k) acc = fmaf(attn_s[k], base[k * 256], acc);
    float s = acc, q = acc * acc;
    #pragma unroll
    for (int off = 16; off > 0; off >>= 1) {
        s += __shfl_xor(s, off, 32);
        q += __shfl_xor(q, off, 32);
    }
    int wave = d >> 5;
    if ((d & 31) == 0) { partS[wave] = s; partQ[wave] = q; }
    __syncthreads();
    if (d == 0) {
        float S = 0.f, Q = 0.f;
        #pragma unroll
        for (int w = 0; w < 8; ++w) { S += partS[w]; Q += partQ[w]; }
        float m = S * (1.f / 256.f);
        float v = Q * (1.f / 256.f) - m * m;
        s_mean = m;
        s_rstd = rsqrtf(v + 1e-5f);
    }
    __syncthreads();
    z_out[(size_t)n * 256 + d] = (acc - s_mean) * s_rstd * fng[d] + fnb[d];
    if (d == 0) {
        float lb = vfeat[(size_t)n * 23 + 21];
        float ub = vfeat[(size_t)n * 23 + 22];
        int   t  = vt[n];
        bool is_int = (t == 2);
        bool finite = (fabsf(lb) < 1e18f) && (fabsf(ub) < 1e18f);
        bool mis  = is_int && finite && ((ub - lb) <= 10.0f);
        bool mil  = is_int && !mis;
        float offs = floorf(lb);
        float rng  = ceilf(ub) - offs + 1.0f;
        rng = fminf(fmaxf(rng, 1.0f), 11.0f);
        o_mb[n]  = (t == 1) ? 1.0f : 0.0f;
        o_mis[n] = mis ? 1.0f : 0.0f;
        o_mil[n] = mil ? 1.0f : 0.0f;
        o_off[n] = offs;
        o_rng[n] = (float)(int)rng;
    }
}

__device__ __forceinline__ float gelu_exact(float x) {
    return 0.5f * x * (1.0f + erff(x * 0.70710678118654752f));
}

// ---------------------------------------------------------------------------
// K3: fused head.  32-row tile / block (8 waves of 32).
//   A: LN(z)->bf16 A-fragments in LDS     B: GEMM1(WMMA bf16)+bias+GELU->LDS
//   C: GEMM2(WMMA bf16)+bias+GELU, residual-add into xres
//   D: (x+h) @ Wh.T + bh, head-specific postprocess
// Tiling: wave w owns BOTH 16-row tiles x one 32-col slab (col tiles 2w,2w+1)
// -> each B fragment (2x global_load_b128) is reused by 2 WMMAs, halving the
// global fetch per WMMA vs a 1-row-tile assignment. 256 WMMAs / block / GEMM.
// ---------------------------------------------------------------------------
__global__ __launch_bounds__(256)
void head_kernel(const float* __restrict__ z,
                 const float* __restrict__ ng, const float* __restrict__ nb,
                 const bf16_t* __restrict__ W1p, const float* __restrict__ b1,
                 const bf16_t* __restrict__ W2p, const float* __restrict__ b2,
                 const float* __restrict__ Wh, const float* __restrict__ bh,
                 float* __restrict__ out, const float* __restrict__ rng,
                 int N, int OUTC, int MODE) {
    __shared__ float xres[M_TILE][256];                        // 32 KB residual
    __shared__ __align__(32) bf16_t Apk[2 * 8 * 32 * 16];      // 16 KB A frags
    __shared__ __align__(32) bf16_t Hpk[2 * 8 * 32 * 16];      // 16 KB h1 frags
    int tid  = threadIdx.x;
    int lane = tid & 31;
    int wave = tid >> 5;
    int row0 = blockIdx.x * M_TILE;

    // ---- Phase A: per-head LayerNorm -> packed bf16 A fragments ----
    for (int rr = 0; rr < 4; ++rr) {
        int rloc = wave * 4 + rr;
        int n = row0 + rloc;
        float vals[8];
        float s = 0.f, q = 0.f;
        #pragma unroll
        for (int u = 0; u < 8; ++u) {
            int c = lane + 32 * u;
            float v = (n < N) ? z[(size_t)n * 256 + c] : 0.f;
            vals[u] = v; s += v; q += v * v;
        }
        #pragma unroll
        for (int off = 16; off > 0; off >>= 1) {
            s += __shfl_xor(s, off, 32);
            q += __shfl_xor(q, off, 32);
        }
        float m  = s * (1.f / 256.f);
        float vv = q * (1.f / 256.f) - m * m;
        float rs = rsqrtf(vv + 1e-5f);
        #pragma unroll
        for (int u = 0; u < 8; ++u) {
            int c = lane + 32 * u;
            xres[rloc][c] = vals[u];
            float ln = (vals[u] - m) * rs * ng[c] + nb[c];
            int rt   = rloc >> 4;
            int kt   = c >> 5;
            int kk   = c & 31;
            int bsel = (kk >> 3) & 1;
            int i    = (kk & 7) + ((kk & 16) ? 8 : 0);
            int l2   = (rloc & 15) + (bsel << 4);
            Apk[(((rt * 8 + kt) * 32) + l2) * 16 + i] = (bf16_t)ln;
        }
    }
    __syncthreads();

    // ---- Phase B: h1 = gelu(LN(z) @ W1.T + b1) ----
    {
        v8f acc[2][2];                 // [row tile][col tile]
        #pragma unroll
        for (int ti = 0; ti < 2; ++ti) {
            float bias = b1[(2 * wave + ti) * 16 + (lane & 15)];
            v8f bv = (v8f){bias, bias, bias, bias, bias, bias, bias, bias};
            acc[0][ti] = bv;
            acc[1][ti] = bv;
        }
        #pragma unroll
        for (int kt = 0; kt < 8; ++kt) {
            v16bf a0 = *(const v16bf*)&Apk[(((0 * 8 + kt) * 32) + lane) * 16];
            v16bf a1 = *(const v16bf*)&Apk[(((1 * 8 + kt) * 32) + lane) * 16];
            #pragma unroll
            for (int ti = 0; ti < 2; ++ti) {
                int nt = 2 * wave + ti;
                v16bf b = *(const v16bf*)&W1p[(size_t)(((nt * 8 + kt) * 32) + lane) * 16];
                acc[0][ti] = __builtin_amdgcn_wmma_f32_16x16x32_bf16(
                    false, a0, false, b, (short)0, acc[0][ti], false, false);
                acc[1][ti] = __builtin_amdgcn_wmma_f32_16x16x32_bf16(
                    false, a1, false, b, (short)0, acc[1][ti], false, false);
            }
        }
        #pragma unroll
        for (int rt = 0; rt < 2; ++rt) {
            #pragma unroll
            for (int ti = 0; ti < 2; ++ti) {
                int c    = (2 * wave + ti) * 16 + (lane & 15);
                int kt   = c >> 5;
                int kk   = c & 31;
                int bsel = (kk >> 3) & 1;
                int i    = (kk & 7) + ((kk & 16) ? 8 : 0);
                #pragma unroll
                for (int v = 0; v < 8; ++v) {
                    int mloc = rt * 16 + v + ((lane >> 4) << 3);   // D layout row
                    float g  = gelu_exact(acc[rt][ti][v]);
                    int l2   = (mloc & 15) + (bsel << 4);
                    int rt2  = mloc >> 4;
                    Hpk[(((rt2 * 8 + kt) * 32) + l2) * 16 + i] = (bf16_t)g;
                }
            }
        }
    }
    __syncthreads();

    // ---- Phase C: h2 = gelu(h1 @ W2.T + b2); xres += h2 ----
    {
        v8f acc[2][2];
        #pragma unroll
        for (int ti = 0; ti < 2; ++ti) {
            float bias = b2[(2 * wave + ti) * 16 + (lane & 15)];
            v8f bv = (v8f){bias, bias, bias, bias, bias, bias, bias, bias};
            acc[0][ti] = bv;
            acc[1][ti] = bv;
        }
        #pragma unroll
        for (int kt = 0; kt < 8; ++kt) {
            v16bf a0 = *(const v16bf*)&Hpk[(((0 * 8 + kt) * 32) + lane) * 16];
            v16bf a1 = *(const v16bf*)&Hpk[(((1 * 8 + kt) * 32) + lane) * 16];
            #pragma unroll
            for (int ti = 0; ti < 2; ++ti) {
                int nt = 2 * wave + ti;
                v16bf b = *(const v16bf*)&W2p[(size_t)(((nt * 8 + kt) * 32) + lane) * 16];
                acc[0][ti] = __builtin_amdgcn_wmma_f32_16x16x32_bf16(
                    false, a0, false, b, (short)0, acc[0][ti], false, false);
                acc[1][ti] = __builtin_amdgcn_wmma_f32_16x16x32_bf16(
                    false, a1, false, b, (short)0, acc[1][ti], false, false);
            }
        }
        #pragma unroll
        for (int rt = 0; rt < 2; ++rt) {
            #pragma unroll
            for (int ti = 0; ti < 2; ++ti) {
                int c = (2 * wave + ti) * 16 + (lane & 15);
                #pragma unroll
                for (int v = 0; v < 8; ++v) {
                    int mloc = rt * 16 + v + ((lane >> 4) << 3);
                    xres[mloc][c] += gelu_exact(acc[rt][ti][v]);   // disjoint (m,c)
                }
            }
        }
    }
    __syncthreads();

    // ---- Phase D: out = (x + h) @ Wh.T + bh (+ head postprocess) ----
    for (int p = tid; p < M_TILE * OUTC; p += 256) {
        int o    = p >> 5;        // output channel
        int mloc = p & 31;
        int n = row0 + mloc;
        if (n >= N) continue;
        float acc = bh[o];
        #pragma unroll 4
        for (int c = 0; c < 256; ++c) acc = fmaf(xres[mloc][c], Wh[o * 256 + c], acc);
        float r;
        if (MODE == 0)      r = 1.f / (1.f + expf(-acc));                      // bin: sigmoid
        else if (MODE == 1) { int R = (int)rng[n]; r = (o < R) ? acc : -1e9f; } // ints: mask
        else                r = acc;                                           // intl
        out[(size_t)n * OUTC + o] = r;
    }
}

// ---------------------------------------------------------------------------
extern "C" void kernel_launch(void* const* d_in, const int* in_sizes, int n_in,
                              void* d_out, int out_size, void* d_ws, size_t ws_size,
                              hipStream_t stream) {
    const int N = in_sizes[1];                       // var_batch: N ints

    const float* evolved   = (const float*)d_in[0];
    const int*   var_batch = (const int*)  d_in[1];
    const float* attn      = (const float*)d_in[2];
    const int*   var_types = (const int*)  d_in[3];
    const float* z_var_0   = (const float*)d_in[4];
    const float* vfeat     = (const float*)d_in[5];
    const float* Wp        = (const float*)d_in[6];
    const float* bp        = (const float*)d_in[7];
    const float* fn_g      = (const float*)d_in[8];
    const float* fn_b      = (const float*)d_in[9];

    // Output tuple, concatenated flat as floats:
    float* out   = (float*)d_out;
    float* o_z   = out;                          // (N,256)
    float* o_pb  = out + (size_t)256 * N;        // (N,1)
    float* o_lg  = out + (size_t)257 * N;        // (N,11)
    float* o_pl  = out + (size_t)268 * N;        // (N,1)
    float* o_mb  = out + (size_t)269 * N;        // (N,)
    float* o_mis = out + (size_t)270 * N;        // (N,)
    float* o_mil = out + (size_t)271 * N;        // (N,)
    float* o_off = out + (size_t)272 * N;        // (N,)
    float* o_rng = out + (size_t)273 * N;        // (N,)

    // Workspace: tok3 (256KB f32) + 6 packed bf16 weights (128KB each)
    char*   ws   = (char*)d_ws;
    float*  tok3 = (float*)ws;
    bf16_t* wpk  = (bf16_t*)(ws + 262144);

    for (int h = 0; h < 3; ++h) {
        const float* W1 = (const float*)d_in[10 + 8 * h + 2];
        const float* W2 = (const float*)d_in[10 + 8 * h + 4];
        pack_weight_bf16<<<256, 256, 0, stream>>>(W1, wpk + (size_t)(2 * h + 0) * 65536);
        pack_weight_bf16<<<256, 256, 0, stream>>>(W2, wpk + (size_t)(2 * h + 1) * 65536);
    }

    proj_kernel<<<256, 256, 0, stream>>>(evolved, Wp, bp, tok3);

    gather_ln_kernel<<<N, 256, 0, stream>>>(var_batch, attn, var_types, z_var_0, vfeat,
                                            tok3, fn_g, fn_b, o_z,
                                            o_mb, o_mis, o_mil, o_off, o_rng, N);

    int blocks = (N + M_TILE - 1) / M_TILE;
    for (int h = 0; h < 3; ++h) {
        const float* ng = (const float*)d_in[10 + 8 * h + 0];
        const float* nb = (const float*)d_in[10 + 8 * h + 1];
        const float* b1 = (const float*)d_in[10 + 8 * h + 3];
        const float* b2 = (const float*)d_in[10 + 8 * h + 5];
        const float* Wh = (const float*)d_in[10 + 8 * h + 6];
        const float* bh = (const float*)d_in[10 + 8 * h + 7];
        float* hout = (h == 0) ? o_pb : (h == 1) ? o_lg : o_pl;
        int outc    = (h == 1) ? 11 : 1;
        head_kernel<<<blocks, 256, 0, stream>>>(o_z, ng, nb,
            wpk + (size_t)(2 * h + 0) * 65536, b1,
            wpk + (size_t)(2 * h + 1) * 65536, b2,
            Wh, bh, hout, o_rng, N, outc, h);
    }
}
